// QMPatchSegmModel_27908697489860
// MI455X (gfx1250) — compile-verified, matrix operand order checked by reading, not run on hardware
//
#include <hip/hip_runtime.h>
#include <hip/hip_bf16.h>

// ---- problem constants (from reference) ----
#define IMGW   96
#define PATCH  4
#define STRIDE 2
#define NPP    47                 // (96-4)/2+1
#define NPAT   (NPP * NPP)        // 2209 patches per image
#define DIMX   48                 // 4*4*3
#define DIMK   64                 // K padded to 64 for WMMA (2 k-steps of 32)
#define NCOMP  64
#define NCLS   12
#define TILES  ((NPAT + 15) / 16) // 139 row tiles of 16 patches
#define NWAVE  8
#define EPS    1e-10f
#define LOG2E  1.44269504088896340736f

typedef __attribute__((ext_vector_type(16))) _Float16 v16h;
typedef __attribute__((ext_vector_type(8)))  _Float16 v8h;
typedef __attribute__((ext_vector_type(2)))  __fp16   v2hp;  // matches cvt_pkrtz return
typedef __attribute__((ext_vector_type(8)))  float    v8f;

typedef union { v16h v16; v8h v8[2]; } frag16;

struct __align__(16) LdsBlk {
    _Float16 cxh[NCOMP][DIMK];        // c_x in f16, K padded with zeros (8 KB)
    _Float16 Atile[NWAVE][16][DIMK];  // per-wave A staging (16 KB)
    float    p2part[NWAVE][16][4];    // per (patch, patch-row) |p|^2 partials
    float    p2sum[NWAVE][16];        // per-patch |p|^2 (1e30 marks padded rows)
    float    cx2[NCOMP];              // |c_x_m|^2
    float    yv2[NCOMP][NCLS];        // (c_y / ||c_y||)^2
    float    compsum[NCOMP];          // cross-wave sum of K^2 over patches
    float    outw[NCOMP];
    float    Ssh;
};

__global__ __launch_bounds__(256)
void qm_patch_segm_kernel(const float* __restrict__ images,
                          const float* __restrict__ c_x,
                          const float* __restrict__ c_y,
                          const float* __restrict__ comp_w,
                          const float* __restrict__ sigma,
                          float* __restrict__ out)
{
    __shared__ LdsBlk lds;

    const int tid  = threadIdx.x;
    const int w    = tid >> 5;        // wave id 0..7
    const int lane = tid & 31;        // wave32 lane
    const int h    = lane >> 4;       // lane half (0/1)
    const int mcol = lane & 15;       // row/col within 16-tile
    const int b    = blockIdx.x;      // image index

    const float sg     = sigma[0];
    const float inv_s2 = 1.0f / (sg * sg);        // K^2 = exp(-d2 / sigma^2)
    const float e_scl  = 2.0f * inv_s2 * LOG2E;   // exp2-domain scale on dot
    const float e_neg  = -inv_s2 * LOG2E;         // exp2-domain scale on |p|^2, |c_x|^2
    const float* img   = images + (size_t)b * IMGW * IMGW * 3;

    // ---------------- one-time block staging ----------------
    for (int idx = tid; idx < NCOMP * DIMK; idx += 256) {
        int m = idx >> 6, k = idx & 63;
        lds.cxh[m][k] = (k < DIMX) ? (_Float16)c_x[m * DIMX + k] : (_Float16)0.0f;
    }
    // zero A-tiles once (k=48..63 pad region stays zero forever)
    for (int idx = tid; idx < NWAVE * 16 * DIMK; idx += 256)
        ((_Float16*)lds.Atile)[idx] = (_Float16)0.0f;
    if (tid < NCOMP) {
        float s = 0.f;
        for (int k = 0; k < DIMX; ++k) { float v = c_x[tid * DIMX + k]; s += v * v; }
        lds.cx2[tid] = s;
        float ny = 0.f;
        for (int c = 0; c < NCLS; ++c) { float v = c_y[tid * NCLS + c]; ny += v * v; }
        float inv_ny = 1.0f / ny;     // (c_y/||c_y||)^2 = c_y^2 / ||c_y||^2
        for (int c = 0; c < NCLS; ++c) {
            float v = c_y[tid * NCLS + c];
            lds.yv2[tid][c] = v * v * inv_ny;
        }
        lds.compsum[tid] = 0.f;
    }
    __syncthreads();

    // ---------------- persistent B fragments (ISA 16-bit 16x32 layout) ----------------
    // lane holds column N = ct*16 + mcol; element j<8 -> K = ks*32 + h*8 + j,
    // j>=8 -> K = ks*32 + 16 + h*8 + (j-8): two contiguous 8-half chunks.
    frag16 bf[4][2];
    float  cbias[4];                  // -|c_x|^2 / sigma^2 * log2(e), per column tile
    for (int ct = 0; ct < 4; ++ct) {
        const v8h* brow = (const v8h*)&lds.cxh[ct * 16 + mcol][0];
        for (int ks = 0; ks < 2; ++ks) {
            bf[ct][ks].v8[0] = brow[ks * 4 + h];
            bf[ct][ks].v8[1] = brow[ks * 4 + 2 + h];
        }
        cbias[ct] = lds.cx2[ct * 16 + mcol] * e_neg;
    }

    float acc[4] = {0.f, 0.f, 0.f, 0.f};

    // ---------------- main loop: each wave owns independent row tiles ----------------
    for (int t = w; t < TILES; t += NWAVE) {
        const int pbase = t * 16;

        // prefetch start of next tile's image rows
        int tn = t + NWAVE;
        if (tn < TILES) {
            int gp2 = tn * 16;
            int py2 = gp2 / NPP, px2 = gp2 - py2 * NPP;
            __builtin_prefetch(img + ((py2 * STRIDE) * IMGW + px2 * STRIDE) * 3, 0, 1);
        }

        // stage 16 patches x 48 feats: 64 (patch, patch-row) segments of 12 floats,
        // packed-pair f16 stores (ds_store_b32)
        #pragma unroll
        for (int seg = 0; seg < 2; ++seg) {
            const int s2 = lane + seg * 32;
            const int pl = s2 >> 2;       // local patch 0..15
            const int pi = s2 & 3;        // patch row 0..3
            const int gp = pbase + pl;    // global patch index
            _Float16* dst = &lds.Atile[w][pl][pi * 12];
            if (gp < NPAT) {
                const int py = gp / NPP;
                const int px = gp - py * NPP;
                // 12 contiguous floats, base is 8-byte aligned -> float2 x6
                const float2* src =
                    (const float2*)(img + ((py * STRIDE + pi) * IMGW + px * STRIDE) * 3);
                float s = 0.f;
                #pragma unroll
                for (int j = 0; j < 6; ++j) {
                    float2 v = src[j];
                    *(v2hp*)(dst + 2 * j) = __builtin_amdgcn_cvt_pkrtz(v.x, v.y);
                    s = __builtin_fmaf(v.x, v.x, s);
                    s = __builtin_fmaf(v.y, v.y, s);
                }
                lds.p2part[w][pl][pi] = s;
            } else {
                const v2hp zz = {(__fp16)0.0f, (__fp16)0.0f};
                #pragma unroll
                for (int j = 0; j < 6; ++j) *(v2hp*)(dst + 2 * j) = zz;
                lds.p2part[w][pl][pi] = (pi == 0) ? 1e30f : 0.f;  // mask pad rows
            }
        }
        if (lane < 16) {
            float4 q = *(const float4*)&lds.p2part[w][lane][0];   // ds_load_b128
            lds.p2sum[w][lane] = (q.x + q.y) + (q.z + q.w);
        }

        // A fragments: lane holds row M = mcol, same chunk pattern as B
        const v8h* arow = (const v8h*)&lds.Atile[w][mcol][0];
        frag16 a0, a1;
        a0.v8[0] = arow[h];     a0.v8[1] = arow[2 + h];
        a1.v8[0] = arow[4 + h]; a1.v8[1] = arow[6 + h];

        // per-row exp2-domain bias: -|p|^2 / sigma^2 * log2(e)
        float4 p2a = *(const float4*)&lds.p2sum[w][8 * h];
        float4 p2b = *(const float4*)&lds.p2sum[w][8 * h + 4];
        float rbias[8];
        rbias[0] = p2a.x * e_neg; rbias[1] = p2a.y * e_neg;
        rbias[2] = p2a.z * e_neg; rbias[3] = p2a.w * e_neg;
        rbias[4] = p2b.x * e_neg; rbias[5] = p2b.y * e_neg;
        rbias[6] = p2b.z * e_neg; rbias[7] = p2b.w * e_neg;

        // 4 column tiles x 2 k-steps of v_wmma_f32_16x16x32_f16
        #pragma unroll
        for (int ct = 0; ct < 4; ++ct) {
            v8f c = {};
            c = __builtin_amdgcn_wmma_f32_16x16x32_f16(false, a0.v16, false, bf[ct][0].v16,
                                                       (short)0, c, false, false);
            c = __builtin_amdgcn_wmma_f32_16x16x32_f16(false, a1.v16, false, bf[ct][1].v16,
                                                       (short)0, c, false, false);
            // fused epilogue: K^2 = exp2(dot*e_scl + rbias + cbias), one fma + one exp each
            float bct[8];
            #pragma unroll
            for (int v = 0; v < 8; ++v) bct[v] = rbias[v] + cbias[ct];
            float s0 = 0.f, s1 = 0.f;
            #pragma unroll
            for (int v = 0; v < 8; v += 2) {
                s0 += __builtin_amdgcn_exp2f(__builtin_fmaf(c[v],     e_scl, bct[v]));
                s1 += __builtin_amdgcn_exp2f(__builtin_fmaf(c[v + 1], e_scl, bct[v + 1]));
            }
            acc[ct] += s0 + s1;
        }
    }

    // ---------------- cross-wave / cross-lane-half reduction ----------------
    #pragma unroll
    for (int ct = 0; ct < 4; ++ct)
        atomicAdd(&lds.compsum[ct * 16 + mcol], acc[ct]);   // ds_add_f32
    __syncthreads();

    if (tid < NCOMP) {
        float wgt = fmaxf(comp_w[tid], 0.f);                 // relu(comp_w)
        lds.outw[tid] = wgt * lds.compsum[tid] * (1.0f / (float)NPAT);  // in_w = 1/n
    }
    __syncthreads();
    if (tid == 0) {
        float s = 0.f;
        for (int m = 0; m < NCOMP; ++m) s += lds.outw[m];
        lds.Ssh = s;
    }
    __syncthreads();
    if (tid < NCLS) {
        float p = 0.f;
        for (int m = 0; m < NCOMP; ++m) p += lds.outw[m] * lds.yv2[m][tid];
        out[b * NCLS + tid] = p / (lds.Ssh + EPS);
    }
}

extern "C" void kernel_launch(void* const* d_in, const int* in_sizes, int n_in,
                              void* d_out, int out_size, void* d_ws, size_t ws_size,
                              hipStream_t stream) {
    const float* images = (const float*)d_in[0];  // (512, 96, 96, 3)
    const float* c_x    = (const float*)d_in[1];  // (64, 48)
    const float* c_y    = (const float*)d_in[2];  // (64, 12)
    const float* comp_w = (const float*)d_in[3];  // (64,)
    const float* sigma  = (const float*)d_in[4];  // scalar
    float* out          = (float*)d_out;          // (512, 12)

    const int B = in_sizes[0] / (IMGW * IMGW * 3);
    qm_patch_segm_kernel<<<B, 256, 0, stream>>>(images, c_x, c_y, comp_w, sigma, out);
}
